// TreeLSTM_13855564497595
// MI455X (gfx1250) — compile-verified, matrix operand order checked by reading, not run on hardware
//
#include <hip/hip_runtime.h>

typedef _Float16 f16;
typedef __attribute__((ext_vector_type(16))) _Float16 v16h;
typedef __attribute__((ext_vector_type(8)))  _Float16 v8h;
typedef __attribute__((ext_vector_type(8)))  float    v8f;
typedef __attribute__((ext_vector_type(4)))  float    v4f;

#define HS   256          // hidden size
#define GS   1024         // 4*H gates
#define KD   256          // K dimension of each GEMM half
#define MT   16           // nodes per block (M tile)
#define NKB  8            // K blocks of 32 (256/32)

__device__ __forceinline__ float sigf(float x) {
    return 1.0f / (1.0f + __expf(-x));
}

// ---------------------------------------------------------------------------
// Pre-pack W (f32, [1024][512] row-major) into two f16 banks in the exact
// per-lane V_WMMA_F32_16X16X32_F16 B-operand layout:
//   Bpack[(kb*64 + gblk)*32 + lane][16 halves]
//   lane: n = lane&15, kbase = kb*32 + (lane&16 ? 8 : 0)
//   halves 0..7  -> K = kbase+0..7
//   halves 8..15 -> K = kbase+16..23
// part 0 (Bh): W columns 0..255  (child_h_sum half)
// part 1 (Bx): W columns 256..511 (x half)
// ---------------------------------------------------------------------------
__global__ void prep_w(const float* __restrict__ W,
                       f16* __restrict__ Bh, f16* __restrict__ Bx) {
    int tid   = blockIdx.x * blockDim.x + threadIdx.x;  // 0..32767
    int part  = tid >> 14;
    int rem   = tid & 16383;
    int kb    = rem >> 11;          // 0..7
    int gblk  = (rem >> 5) & 63;    // 0..63
    int lane  = rem & 31;
    int g     = gblk * 16 + (lane & 15);
    int kbase = kb * 32 + ((lane & 16) ? 8 : 0);
    const float* wrow = W + (size_t)g * 512 + part * 256;
    v16h frag;
#pragma unroll
    for (int h = 0; h < 8; ++h) {
        frag[h]     = (f16)wrow[kbase + h];
        frag[8 + h] = (f16)wrow[kbase + 16 + h];
    }
    f16* dst = (part ? Bx : Bh) + ((size_t)((kb * 64 + gblk) * 32 + lane) << 4);
    *(v16h*)dst = frag;
}

// ---------------------------------------------------------------------------
// One tree level: gates = A @ Bpack (+bias), LSTM cell fusion, store h(f16)/c(f32).
// mode 0: A = leaf features (f32->f16). mode 1: A = h_prev[2n] + h_prev[2n+1].
// Block = 256 threads (8 wave32), 16 nodes. Wave w owns gate columns
// {gb*256 + w*32 + s*16 + [0..15] : gb in 0..3, s in 0..1} so f/i/c~/o fuse
// lane-locally.
// ---------------------------------------------------------------------------
__global__ __launch_bounds__(256) void tree_level(
    int mode,
    const float* __restrict__ x,       // leaf features (mode 0)
    const f16*   __restrict__ h_prev,  // mode 1
    const float* __restrict__ c_prev,  // mode 1
    const f16*   __restrict__ Bpack,
    const float* __restrict__ bias,
    f16*   __restrict__ h_out,
    float* __restrict__ c_out,
    int n_nodes,
    float* __restrict__ final_out) {

    __shared__ __align__(32) f16 Atile[NKB][32][16];   // 8 KB, pre-swizzled A frags

    const int tid   = threadIdx.x;
    const int node0 = blockIdx.x * MT;

    // ---- stage A tile into LDS in per-lane WMMA A-fragment order ----
    {
        int kb    = tid >> 5;                       // 0..7
        int slot  = tid & 31;                       // fragment lane slot
        int m     = slot & 15;
        int node  = node0 + m;
        int kbase = kb * 32 + ((slot & 16) ? 8 : 0);
        v16h frag;
        if (node < n_nodes) {
            if (mode == 0) {
                const float* src = x + (size_t)node * KD;
                v4f q0 = *(const v4f*)(src + kbase);
                v4f q1 = *(const v4f*)(src + kbase + 4);
                v4f q2 = *(const v4f*)(src + kbase + 16);
                v4f q3 = *(const v4f*)(src + kbase + 20);
#pragma unroll
                for (int h = 0; h < 4; ++h) {
                    frag[h]      = (f16)q0[h];
                    frag[4 + h]  = (f16)q1[h];
                    frag[8 + h]  = (f16)q2[h];
                    frag[12 + h] = (f16)q3[h];
                }
            } else {
                const f16* s0 = h_prev + (size_t)(2 * node) * HS;
                const f16* s1 = h_prev + (size_t)(2 * node + 1) * HS;
                v8h a0 = *(const v8h*)(s0 + kbase);
                v8h b0 = *(const v8h*)(s1 + kbase);
                v8h a1 = *(const v8h*)(s0 + kbase + 16);
                v8h b1 = *(const v8h*)(s1 + kbase + 16);
#pragma unroll
                for (int h = 0; h < 8; ++h) {
                    frag[h]     = (f16)(a0[h] + b0[h]);
                    frag[8 + h] = (f16)(a1[h] + b1[h]);
                }
            }
        } else {
#pragma unroll
            for (int h = 0; h < 16; ++h) frag[h] = (f16)0.0f;
        }
        *(v16h*)&Atile[kb][slot][0] = frag;
    }
    __syncthreads();

    const int wave = tid >> 5;
    const int lane = tid & 31;

    v8f zero = {};
    v8f acc[4][2];
#pragma unroll
    for (int gb = 0; gb < 4; ++gb)
#pragma unroll
        for (int s = 0; s < 2; ++s) acc[gb][s] = zero;

    // ---- K loop: 8 k-blocks x (4 gates x 2 n-subtiles) WMMAs ----
#pragma unroll
    for (int kb = 0; kb < NKB; ++kb) {
        v16h afrag = *(const v16h*)&Atile[kb][lane][0];
        if (kb + 1 < NKB) {
            int gblk_n = wave * 2;   // first frag of next k-block for this wave
            __builtin_prefetch(
                Bpack + ((size_t)(((kb + 1) * 64 + gblk_n) * 32 + lane) << 4), 0, 1);
        }
#pragma unroll
        for (int gb = 0; gb < 4; ++gb) {
#pragma unroll
            for (int s = 0; s < 2; ++s) {
                int gblk = gb * 16 + wave * 2 + s;
                v16h bfrag = *(const v16h*)(
                    Bpack + ((size_t)((kb * 64 + gblk) * 32 + lane) << 4));
                acc[gb][s] = __builtin_amdgcn_wmma_f32_16x16x32_f16(
                    false, afrag, false, bfrag, (short)0, acc[gb][s], false, false);
            }
        }
    }

    // ---- LSTM cell fusion + stores ----
    const int lhalf = lane & 15;
    const int lhi   = lane >> 4;
#pragma unroll
    for (int s = 0; s < 2; ++s) {
        int hcol = wave * 32 + s * 16 + lhalf;      // 0..255
        float bf = bias[hcol];
        float bi = bias[HS + hcol];
        float bc = bias[2 * HS + hcol];
        float bo = bias[3 * HS + hcol];
#pragma unroll
        for (int v = 0; v < 8; ++v) {
            int m    = v + lhi * 8;                 // C/D layout: VGPR v, lane-half
            int node = node0 + m;
            if (node < n_nodes) {
                float gf = sigf(acc[0][s][v] + bf);
                float gi = sigf(acc[1][s][v] + bi);
                float gc = tanhf(acc[2][s][v] + bc);
                float go = sigf(acc[3][s][v] + bo);
                float csum = 0.0f;
                if (mode) {
                    csum = c_prev[(size_t)(2 * node) * HS + hcol] +
                           c_prev[(size_t)(2 * node + 1) * HS + hcol];
                }
                float c = gf * csum + gi * gc;
                float h = go * tanhf(c);
                h_out[(size_t)node * HS + hcol] = (f16)h;
                c_out[(size_t)node * HS + hcol] = c;
                if (final_out) final_out[hcol] = h;   // n==1: node 0 covers all cols
            }
        }
    }
}

// ---------------------------------------------------------------------------
// Host launch: prep weights once, then walk 18 dependent levels (leaf + 17).
// Workspace: [Bh 512K][Bx 512K][hA 64M][hB 64M][cA 128M][cB 128M] = ~385 MB.
// ---------------------------------------------------------------------------
extern "C" void kernel_launch(void* const* d_in, const int* in_sizes, int n_in,
                              void* d_out, int out_size, void* d_ws, size_t ws_size,
                              hipStream_t stream) {
    const float* leaf = (const float*)d_in[0];
    const float* W    = (const float*)d_in[1];
    const float* b    = (const float*)d_in[2];

    char* ws = (char*)d_ws;
    f16*   Bh = (f16*)ws;
    f16*   Bx = Bh + (size_t)8 * 64 * 32 * 16;                 // +512 KB
    f16*   hA = (f16*)(ws + ((size_t)1 << 20));                // 64 MB
    f16*   hB = hA + (size_t)131072 * HS;                      // 64 MB
    float* cA = (float*)(ws + ((size_t)1 << 20) + ((size_t)128 << 20)); // 128 MB
    float* cB = cA + (size_t)131072 * HS;                      // 128 MB

    prep_w<<<128, 256, 0, stream>>>(W, Bh, Bx);

    int n = 131072;
    tree_level<<<n / MT, 256, 0, stream>>>(0, leaf, nullptr, nullptr,
                                           Bx, b, hA, cA, n, nullptr);
    f16*   hs = hA;  float* cs = cA;
    f16*   hd = hB;  float* cd = cB;
    while (n > 1) {
        n >>= 1;
        float* fin = (n == 1) ? (float*)d_out : nullptr;
        tree_level<<<(n + MT - 1) / MT, 256, 0, stream>>>(1, nullptr, hs, cs,
                                                          Bh, b, hd, cd, n, fin);
        f16* th = hs; hs = hd; hd = th;
        float* tc = cs; cs = cd; cd = tc;
    }
}
